// MPNNRegressor_91036126806070
// MI455X (gfx1250) — matrix operand order repaired and checked by reading, not compiled, not run on hardware
//
#include <hip/hip_runtime.h>
#include <hip/hip_bf16.h>
#include <stdint.h>

// ---------------------------------------------------------------------------
// MPNN regressor forward for gfx1250 (MI455X), bf16 WMMA + fp32 accumulation.
// ---------------------------------------------------------------------------

typedef __attribute__((ext_vector_type(16))) __bf16 v16bf;
typedef __attribute__((ext_vector_type(8)))  float  v8f;

#define DD 128          // feature dim
#define LDSPITCH 132    // words per LDS row (264 bf16 = 256 data + 8 pad)

__device__ __forceinline__ unsigned pack2bf(float lo, float hi) {
  union { __bf16 h[2]; unsigned u; } p;
  p.h[0] = (__bf16)lo; p.h[1] = (__bf16)hi;
  return p.u;
}

__device__ __forceinline__ v16bf ldsfrag(const unsigned* w, int off0, int off1) {
  union { uint4 q[2]; v16bf v; } u;
  u.q[0] = *(const uint4*)(w + off0);   // ds_load_b128
  u.q[1] = *(const uint4*)(w + off1);   // ds_load_b128
  return u.v;
}

__device__ __forceinline__ v16bf pack16(const float* f) {
  v16bf v;
#pragma unroll
  for (int i = 0; i < 16; ++i) v[i] = (__bf16)f[i];
  return v;
}

// ---------------------------------------------------------------------------
// Fused: gather x[row]/x[col] -> LDS(bf16) ; [16e x 256] @ W^T -> +b, ReLU ;
// atomic scatter-add into agg[row].  One WG = 128 edges = 8 waves x 16 edges.
// ---------------------------------------------------------------------------
__global__ __launch_bounds__(256)
void msg_kernel(const float* __restrict__ x,
                const long long* __restrict__ row,
                const long long* __restrict__ col,
                const float* __restrict__ Wm,   // [128][256] row-major
                const float* __restrict__ bm,   // [128]
                float* __restrict__ agg,        // [N][128], atomically updated
                int E_) {
  extern __shared__ char smem[];
  unsigned* Aw   = (unsigned*)smem;               // 128 rows x 132 words (A, bf16)
  unsigned* Ww   = Aw + 128 * LDSPITCH;           // 128 rows x 132 words (W, bf16)
  int*      erow = (int*)(Ww + 128 * LDSPITCH);   // 128 destination node ids

  const int t     = threadIdx.x;
  const int ebase = blockIdx.x * 128;

  // ---- stage A (gathered, bf16) and W (bf16) into LDS --------------------
  {
    const int el = t >> 1, h = t & 1;
    int eg = ebase + el; if (eg >= E_) eg = E_ - 1;
    const long long src = h ? col[eg] : row[eg];
    if (h == 0) erow[el] = (int)src;
    const float4* xs = (const float4*)(x + (size_t)src * DD);
    unsigned* ad = Aw + el * LDSPITCH + h * 64;
    const float4* wsrc = (const float4*)(Wm + (size_t)el * 256 + h * 128);
    unsigned* wd = Ww + el * LDSPITCH + h * 64;
#pragma unroll 8
    for (int i = 0; i < 32; ++i) {
      float4 f = xs[i];
      ad[2 * i]     = pack2bf(f.x, f.y);
      ad[2 * i + 1] = pack2bf(f.z, f.w);
      float4 g = wsrc[i];
      wd[2 * i]     = pack2bf(g.x, g.y);
      wd[2 * i + 1] = pack2bf(g.z, g.w);
    }
  }
  __syncthreads();

  const int wv = t >> 5, lane = t & 31;
  const int lr = lane & 15, hb = lane >> 4;

  v8f acc[8];
#pragma unroll
  for (int i = 0; i < 8; ++i)
#pragma unroll
    for (int j = 0; j < 8; ++j) acc[i][j] = 0.f;

  // ---- 16 edges x 128 outputs, K = 256 : 8 ksteps x 8 ntiles WMMA --------
#pragma unroll
  for (int ks = 0; ks < 8; ++ks) {
    const int aoff = (wv * 16 + lr) * LDSPITCH + ks * 16 + hb * 4;
    v16bf af = ldsfrag(Aw, aoff, aoff + 8);
#pragma unroll
    for (int nt = 0; nt < 8; ++nt) {
      const int boff = (nt * 16 + lr) * LDSPITCH + ks * 16 + hb * 4;
      v16bf bf = ldsfrag(Ww, boff, boff + 8);
      acc[nt] = __builtin_amdgcn_wmma_f32_16x16x32_bf16(
          false, af, false, bf, (short)0, acc[nt], false, false);
    }
  }

  // ---- +bias, ReLU, atomic scatter-add to agg[row[e]] --------------------
#pragma unroll
  for (int nt = 0; nt < 8; ++nt) {
    const int n = nt * 16 + lr;
    const float b = bm[n];
#pragma unroll
    for (int r = 0; r < 8; ++r) {
      const int m = wv * 16 + r + 8 * hb;          // edge slot within block
      if (ebase + m < E_) {
        float v = acc[nt][r] + b;
        v = v > 0.f ? v : 0.f;
        atomicAdd(agg + (size_t)erow[m] * DD + n, v);   // global_atomic_add_f32
      }
    }
  }
}

// ---------------------------------------------------------------------------
// out[M, Nout] = A[M,128] @ W[Nout,128]^T + bias.  A-fragments cached in
// registers across all N-tiles; W fragments streamed from L2, cvt to bf16.
// ---------------------------------------------------------------------------
__global__ __launch_bounds__(256)
void gemm_bias(const float* __restrict__ A, const float* __restrict__ W,
               const float* __restrict__ bias, float* __restrict__ out,
               int M, int Nout) {
  const int t = threadIdx.x, wv = t >> 5, lane = t & 31;
  const int lr = lane & 15, hb = lane >> 4;
  const int mrow = (blockIdx.x * 8 + wv) * 16;
  if (mrow >= M) return;                         // wave-uniform exit
  int ar = mrow + lr; if (ar >= M) ar = M - 1;   // clamp tail rows for loads
  const float* Abase = A + (size_t)ar * DD;

  v16bf afr[4];
#pragma unroll
  for (int ks = 0; ks < 4; ++ks) {
    const float* p = Abase + ks * 32 + hb * 8;
    float f[16];
    *(float4*)(f)      = *(const float4*)(p);
    *(float4*)(f + 4)  = *(const float4*)(p + 4);
    *(float4*)(f + 8)  = *(const float4*)(p + 16);
    *(float4*)(f + 12) = *(const float4*)(p + 20);
    afr[ks] = pack16(f);
  }

  const int ntiles = Nout >> 4;
  for (int nt = 0; nt < ntiles; ++nt) {
    const int n = nt * 16 + lr;
    const float* Wr = W + (size_t)n * DD;
    v8f acc;
#pragma unroll
    for (int j = 0; j < 8; ++j) acc[j] = 0.f;
#pragma unroll
    for (int ks = 0; ks < 4; ++ks) {
      const float* p = Wr + ks * 32 + hb * 8;
      float g[16];
      *(float4*)(g)      = *(const float4*)(p);
      *(float4*)(g + 4)  = *(const float4*)(p + 4);
      *(float4*)(g + 8)  = *(const float4*)(p + 16);
      *(float4*)(g + 12) = *(const float4*)(p + 20);
      acc = __builtin_amdgcn_wmma_f32_16x16x32_bf16(
          false, afr[ks], false, pack16(g), (short)0, acc, false, false);
    }
    const float b = bias[n];
#pragma unroll
    for (int r = 0; r < 8; ++r) {
      const int m = mrow + r + 8 * hb;
      if (m < M) out[(size_t)m * Nout + n] = acc[r] + b;
    }
  }
}

// ---------------------------------------------------------------------------
__global__ void gru_elem(const float* __restrict__ gi, const float* __restrict__ gh,
                         float* __restrict__ xb, int N_) {
  const int idx = blockIdx.x * blockDim.x + threadIdx.x;
  if (idx >= N_ * DD) return;
  const int n = idx >> 7, d = idx & (DD - 1);
  const size_t gb = (size_t)n * 384 + d;
  const float ir = gi[gb], iz = gi[gb + 128], in_ = gi[gb + 256];
  const float hr = gh[gb], hz = gh[gb + 128], hn  = gh[gb + 256];
  const float h = xb[idx];
  const float r = 1.f / (1.f + __expf(-(ir + hr)));
  const float z = 1.f / (1.f + __expf(-(iz + hz)));
  const float nn = tanhf(in_ + r * hn);
  xb[idx] = (1.f - z) * nn + z * h;
}

__global__ void copy4(const float4* __restrict__ src, float4* __restrict__ dst, int n4) {
  const int i = blockIdx.x * blockDim.x + threadIdx.x;
  if (i < n4) dst[i] = src[i];
}

__global__ void fill0(float* __restrict__ p, int n) {
  const int i = blockIdx.x * blockDim.x + threadIdx.x;
  if (i < n) p[i] = 0.f;
}

__global__ void pool_kernel(const float* __restrict__ xb, const long long* __restrict__ batch,
                            float* __restrict__ sums, float* __restrict__ cnt, int N_) {
  const int idx = blockIdx.x * blockDim.x + threadIdx.x;
  if (idx >= N_ * DD) return;
  const int n = idx >> 7, d = idx & (DD - 1);
  const int g = (int)batch[n];
  atomicAdd(sums + (size_t)g * DD + d, xb[idx]);
  if (d == 0) atomicAdd(cnt + g, 1.f);
}

__global__ void readout(const float* __restrict__ sums, const float* __restrict__ cnt,
                        const float* __restrict__ fcW, const float* __restrict__ fcb,
                        float* __restrict__ out) {
  __shared__ float red[DD];
  const int g = blockIdx.x, t = threadIdx.x;
  const float c = fmaxf(cnt[g], 1.f);
  red[t] = sums[(size_t)g * DD + t] / c * fcW[t];
  __syncthreads();
  for (int s = 64; s > 0; s >>= 1) {
    if (t < s) red[t] += red[t + s];
    __syncthreads();
  }
  if (t == 0) out[g] = red[0] + fcb[0];
}

// ---------------------------------------------------------------------------
extern "C" void kernel_launch(void* const* d_in, const int* in_sizes, int n_in,
                              void* d_out, int out_size, void* d_ws, size_t ws_size,
                              hipStream_t stream) {
  (void)n_in; (void)ws_size;
  const float*     x     = (const float*)d_in[0];
  const long long* ei    = (const long long*)d_in[1];   // int64 [2, E]
  const long long* batch = (const long long*)d_in[2];   // int64 [N]
  const float* msgW = (const float*)d_in[3];            // [L,128,256]
  const float* msgb = (const float*)d_in[4];            // [L,128]
  const float* Wih  = (const float*)d_in[5];            // [L,384,128]
  const float* Whh  = (const float*)d_in[6];            // [L,384,128]
  const float* bih  = (const float*)d_in[7];            // [L,384]
  const float* bhh  = (const float*)d_in[8];            // [L,384]
  const float* fcW  = (const float*)d_in[9];            // [128]
  const float* fcb  = (const float*)d_in[10];           // [1]
  float* out = (float*)d_out;

  const int N = in_sizes[0] / DD;       // 20000
  const int E = in_sizes[1] / 2;        // 640000
  const int G = out_size;               // 128

  float* xb   = (float*)d_ws;
  float* agg  = xb  + (size_t)N * DD;
  float* gi   = agg + (size_t)N * DD;
  float* gh   = gi  + (size_t)N * 384;
  float* sums = gh  + (size_t)N * 384;
  float* cnt  = sums + (size_t)G * DD;

  const size_t smem = (size_t)2 * 128 * LDSPITCH * 4 + 128 * 4;  // ~135.7 KB
  const int nd = N * DD;

  copy4<<<(nd / 4 + 255) / 256, 256, 0, stream>>>((const float4*)x, (float4*)xb, nd / 4);

  for (int l = 0; l < 3; ++l) {
    fill0<<<(nd + 255) / 256, 256, 0, stream>>>(agg, nd);
    msg_kernel<<<(E + 127) / 128, 256, smem, stream>>>(
        xb, ei, ei + E, msgW + (size_t)l * 128 * 256, msgb + (size_t)l * 128, agg, E);
    const int gb = (N + 127) / 128;
    gemm_bias<<<gb, 256, 0, stream>>>(agg, Wih + (size_t)l * 384 * 128,
                                      bih + (size_t)l * 384, gi, N, 384);
    gemm_bias<<<gb, 256, 0, stream>>>(xb,  Whh + (size_t)l * 384 * 128,
                                      bhh + (size_t)l * 384, gh, N, 384);
    gru_elem<<<(nd + 255) / 256, 256, 0, stream>>>(gi, gh, xb, N);
  }

  fill0<<<(G * DD + G + 255) / 256, 256, 0, stream>>>(sums, G * DD + G);
  pool_kernel<<<(nd + 255) / 256, 256, 0, stream>>>(xb, batch, sums, cnt, N);
  readout<<<G, DD, 0, stream>>>(sums, cnt, fcW, fcb, out);
}
